// BaseAttention_14018773254453
// MI455X (gfx1250) — matrix operand order repaired and checked by reading, not compiled, not run on hardware
//
#include <hip/hip_runtime.h>
#include <hip/hip_bf16.h>

// Problem constants (match reference)
#define BB 2
#define HH 16
#define NN 2048
#define DD 64
#define CC 1024
#define MTOT (BB*NN)        // 4096 tokens total
#define ATT_SCALE 0.125f    // 64^-0.5

typedef __attribute__((ext_vector_type(16))) __bf16 bf16x16;
typedef __attribute__((ext_vector_type(8)))  float  f32x8;
typedef __attribute__((ext_vector_type(4)))  unsigned int u32x4;
typedef __attribute__((ext_vector_type(8)))  int i32x8;
typedef __attribute__((ext_vector_type(4)))  int i32x4;

#define WMMA_BF16(a,b,c) \
  __builtin_amdgcn_wmma_f32_16x16x32_bf16(false,(a),false,(b),(short)0,(c),false,false)

static __device__ inline f32x8 zero8() {
  f32x8 z = {0.f,0.f,0.f,0.f,0.f,0.f,0.f,0.f};
  return z;
}

// ---- WMMA fragment loaders (layouts per CDNA5 ISA 7.12.2, wave32) ----
// A: 16x32 (MxK). lane%16 = M. elem e -> K = e + 8*(lane/16) + (e>=8 ? 8 : 0).
static __device__ inline bf16x16 afrag_f32(const float* base, int ldk, int lane, int kbase) {
  int m = lane & 15, g = lane >> 4;
  const float* row = base + m * ldk + kbase;
  bf16x16 f;
#pragma unroll
  for (int p = 0; p < 8; ++p) {
    int k = 2*p + 8*g + (p >= 4 ? 8 : 0);
    f[2*p]   = (__bf16)row[k];
    f[2*p+1] = (__bf16)row[k+1];
  }
  return f;
}
static __device__ inline bf16x16 afrag_bf16(const __bf16* base, int ldk, int lane, int kbase) {
  int m = lane & 15, g = lane >> 4;
  const __bf16* row = base + m * ldk + kbase;
  bf16x16 f;
#pragma unroll
  for (int p = 0; p < 8; ++p) {
    int k = 2*p + 8*g + (p >= 4 ? 8 : 0);
    f[2*p]   = row[k];
    f[2*p+1] = row[k+1];
  }
  return f;
}
// B: 32x16 (KxN). lane%16 = N, elem e -> K = e + 8*(lane/16) + (e>=8 ? 8 : 0). Source row-major [N][ldk].
static __device__ inline bf16x16 bfrag_bf16(const __bf16* base, int ldk, int lane, int kbase) {
  int n = lane & 15, g = lane >> 4;
  const __bf16* p = base + n * ldk + kbase;
  bf16x16 f;
#pragma unroll
  for (int e = 0; e < 16; ++e) f[e] = p[e + 8 * g + (e >= 8 ? 8 : 0)];
  return f;
}

// ---- Tensor Data Mover: 2D bf16 tile global->LDS (ISA ch.8 D# layout) ----
// group0: [1:0]=count=1, [63:32]=lds_addr, [120:64]=global_addr, [127:126]=type=2
// group1: [17:16]=data_size=1 (2B), [79:48]=tensor_dim0, [111:80]=tensor_dim1,
//         [127:112]=tile_dim0, [143:128]=tile_dim1, [207:160]=tensor_dim0_stride
static __device__ inline unsigned lds_byte_addr(const void* p) {
  // LDS aperture: flat addr[31:0] is the LDS byte offset
  return (unsigned)(unsigned long long)p;
}
static __device__ inline void tdm_load_2d(unsigned lds_addr, const __bf16* gptr,
                                          unsigned tile_d0, unsigned tile_d1,
                                          unsigned tensor_d0, unsigned tensor_d1,
                                          unsigned long long stride0_elems)
{
  unsigned long long ga = (unsigned long long)(size_t)gptr;
  u32x4 g0;
  g0[0] = 1u;                                            // count=1 (valid user desc)
  g0[1] = lds_addr;                                      // LDS byte address
  g0[2] = (unsigned)(ga & 0xffffffffu);                  // global_addr[31:0]
  g0[3] = (unsigned)((ga >> 32) & 0x1ffffffu) | (2u << 30); // ga[56:32] | type=2
  i32x8 g1;
  g1[0] = 0x00010000;                                    // wg_mask=0, data_size=2B
  g1[1] = (int)((tensor_d0 & 0xffffu) << 16);            // tensor_dim0[15:0]
  g1[2] = (int)((tensor_d0 >> 16) | ((tensor_d1 & 0xffffu) << 16));
  g1[3] = (int)((tensor_d1 >> 16) | (tile_d0 << 16));    // tile_dim0
  g1[4] = (int)(tile_d1 & 0xffffu);                      // tile_dim1, tile_dim2=0
  g1[5] = (int)(stride0_elems & 0xffffffffu);            // tensor_dim0_stride lo
  g1[6] = (int)((stride0_elems >> 32) & 0xffffu);        // stride0 hi, stride1=0
  g1[7] = 0;
  i32x4 zz4 = {0, 0, 0, 0};
  i32x8 zz8 = {0, 0, 0, 0, 0, 0, 0, 0};
  // amdgpu-toolchain (clang-23) 6-arg form: g0, g1, g2, g3, extra group, cpol
  __builtin_amdgcn_tensor_load_to_lds(g0, g1, zz4, zz4, zz8, 0);
}

// =====================================================================
// Kernel 1: QKV projections.  X[4096,1024]fp32 @ W[1024,1024]fp32 -> bf16
// z=0: Q -> qh[B,H,N,D]   z=1: K -> kh[B,H,N,D]   z=2: V -> vt[B,H,D,N]
// =====================================================================
__global__ __launch_bounds__(256)
void k_qkv_proj(const float* __restrict__ qi, const float* __restrict__ ki,
                const float* __restrict__ vi,
                const float* __restrict__ Wq, const float* __restrict__ Wk,
                const float* __restrict__ Wv,
                __bf16* __restrict__ qh, __bf16* __restrict__ kh,
                __bf16* __restrict__ vt)
{
  __shared__ __bf16 ldsB[64 * 32];   // W^T tile: [n][k], k contiguous
  const int z = blockIdx.z;
  const float* X = (z == 0) ? qi : (z == 1) ? ki : vi;
  const float* W = (z == 0) ? Wq : (z == 1) ? Wk : Wv;

  const int tid  = threadIdx.x;
  const int lane = tid & 31, wave = tid >> 5;
  const int n0 = blockIdx.x * 64;
  const int m0 = blockIdx.y * 128 + wave * 16;

  f32x8 acc0 = zero8(), acc1 = zero8(), acc2 = zero8(), acc3 = zero8();
  const int nc = tid & 63;
  const int kr = tid >> 6;

  for (int kk = 0; kk < CC; kk += 32) {
    __syncthreads();
#pragma unroll
    for (int i = 0; i < 8; ++i) {
      int krow = kr + i * 4;
      ldsB[nc * 32 + krow] = (__bf16)W[(size_t)(kk + krow) * CC + n0 + nc];
    }
    __syncthreads();
    bf16x16 a  = afrag_f32(X + (size_t)m0 * CC, CC, lane, kk);
    bf16x16 b0 = bfrag_bf16(ldsB + 0 * 16 * 32, 32, lane, 0);
    bf16x16 b1 = bfrag_bf16(ldsB + 1 * 16 * 32, 32, lane, 0);
    bf16x16 b2 = bfrag_bf16(ldsB + 2 * 16 * 32, 32, lane, 0);
    bf16x16 b3 = bfrag_bf16(ldsB + 3 * 16 * 32, 32, lane, 0);
    acc0 = WMMA_BF16(a, b0, acc0);
    acc1 = WMMA_BF16(a, b1, acc1);
    acc2 = WMMA_BF16(a, b2, acc2);
    acc3 = WMMA_BF16(a, b3, acc3);
  }

  const int g = lane >> 4, nl = lane & 15;
#pragma unroll
  for (int r = 0; r < 8; ++r) {
    int mrow = m0 + r + 8 * g;
    int b = mrow / NN, tok = mrow % NN;
#pragma unroll
    for (int sub = 0; sub < 4; ++sub) {
      int c = n0 + sub * 16 + nl;
      int h = c >> 6, d = c & 63;
      float val = (sub == 0) ? acc0[r] : (sub == 1) ? acc1[r]
                : (sub == 2) ? acc2[r] : acc3[r];
      if (z == 2) {
        vt[(((size_t)b * HH + h) * DD + d) * NN + tok] = (__bf16)val;
      } else {
        __bf16* dst = (z == 0) ? qh : kh;
        dst[(((size_t)b * HH + h) * NN + tok) * DD + d] = (__bf16)val;
      }
    }
  }
}

// =====================================================================
// Kernel 2: flash attention with TDM double-buffered K/V tiles.
// Block 128 (4 waves); wave = 16 queries; all waves share (b,h) K/V.
// Wave 0 issues tensor_load_to_lds for next 32-key block while all
// waves compute on the current one (s_wait_tensorcnt + barrier sync).
// =====================================================================
__global__ __launch_bounds__(128)
void k_attn(const __bf16* __restrict__ qh, const __bf16* __restrict__ kh,
            const __bf16* __restrict__ vt, __bf16* __restrict__ ao)
{
  __shared__ __bf16 ldsK[2][32 * 64];   // [keys][d]  (K rows, d contiguous)
  __shared__ __bf16 ldsV[2][64 * 32];   // [d][keys]  (V^T rows, keys contiguous)
  __shared__ __bf16 ldsP[4][16 * 32];   // per-wave P tile

  const int tid  = threadIdx.x;
  const int lane = tid & 31, wave = tid >> 5;
  const int bh = blockIdx.y;
  const int b  = bh >> 4, h = bh & 15;
  const int q0 = blockIdx.x * 64 + wave * 16;

  const __bf16* qbase = qh + ((size_t)bh * NN + q0) * DD;
  const __bf16* kbase = kh + (size_t)bh * NN * DD;
  const __bf16* vbase = vt + (size_t)bh * DD * NN;

  bf16x16 aq0 = afrag_bf16(qbase, DD, lane, 0);
  bf16x16 aq1 = afrag_bf16(qbase, DD, lane, 32);

  float mstate[8], lstate[8];
  f32x8 o0 = zero8(), o1 = zero8(), o2 = zero8(), o3 = zero8();
#pragma unroll
  for (int r = 0; r < 8; ++r) { mstate[r] = -1e30f; lstate[r] = 0.f; }

  __bf16* pl = ldsP[wave];
  const int g = lane >> 4, nl = lane & 15;

  // Prologue: fetch block 0 into buffer 0 (K tile is fully contiguous -> 1D)
  if (wave == 0) {
    tdm_load_2d(lds_byte_addr(ldsK[0]), kbase, 2048u, 1u, 2048u, 1u, 2048ull);
    tdm_load_2d(lds_byte_addr(ldsV[0]), vbase, 32u, 64u, 32u, 64u, (unsigned long long)NN);
  }

  const int NIT = NN / 32;
  for (int it = 0; it < NIT; ++it) {
    const int kb  = it * 32;
    const int buf = it & 1;

    if (wave == 0) __builtin_amdgcn_s_wait_tensorcnt(0);
    __syncthreads();
    if (wave == 0 && it + 1 < NIT) {
      tdm_load_2d(lds_byte_addr(ldsK[buf ^ 1]), kbase + (size_t)(kb + 32) * DD,
                  2048u, 1u, 2048u, 1u, 2048ull);
      tdm_load_2d(lds_byte_addr(ldsV[buf ^ 1]), vbase + (kb + 32),
                  32u, 64u, 32u, 64u, (unsigned long long)NN);
    }

    const __bf16* Kt = ldsK[buf];   // [32 keys][64 d]
    const __bf16* Vt = ldsV[buf];   // [64 d][32 keys]

    // ---- scores: S = (Q @ K^T) * scale ----
    bf16x16 bk00 = bfrag_bf16(Kt,           DD, lane, 0);
    bf16x16 bk01 = bfrag_bf16(Kt,           DD, lane, 32);
    bf16x16 bk10 = bfrag_bf16(Kt + 16 * DD, DD, lane, 0);
    bf16x16 bk11 = bfrag_bf16(Kt + 16 * DD, DD, lane, 32);
    f32x8 s0 = zero8(), s1 = zero8();
    s0 = WMMA_BF16(aq0, bk00, s0);
    s0 = WMMA_BF16(aq1, bk01, s0);
    s1 = WMMA_BF16(aq0, bk10, s1);
    s1 = WMMA_BF16(aq1, bk11, s1);

    // ---- online softmax (row = r + 8*g, cols across 16 lanes of a half) ----
    float corr[8];
#pragma unroll
    for (int r = 0; r < 8; ++r) {
      float a = s0[r] * ATT_SCALE, c = s1[r] * ATT_SCALE;
      float mx = fmaxf(a, c);
#pragma unroll
      for (int t = 1; t < 16; t <<= 1) mx = fmaxf(mx, __shfl_xor(mx, t, 32));
      float mnew = fmaxf(mstate[r], mx);
      float p0 = __expf(a - mnew);
      float p1 = __expf(c - mnew);
      float rs = p0 + p1;
#pragma unroll
      for (int t = 1; t < 16; t <<= 1) rs += __shfl_xor(rs, t, 32);
      corr[r]   = __expf(mstate[r] - mnew);
      lstate[r] = lstate[r] * corr[r] + rs;
      mstate[r] = mnew;
      pl[(r + 8 * g) * 32 + nl]      = (__bf16)p0;
      pl[(r + 8 * g) * 32 + 16 + nl] = (__bf16)p1;
    }
#pragma unroll
    for (int r = 0; r < 8; ++r) {
      o0[r] *= corr[r]; o1[r] *= corr[r]; o2[r] *= corr[r]; o3[r] *= corr[r];
    }

    // same-wave LDS RAW on the P tile (LDS pipe is in-order per wave)
    asm volatile("s_wait_dscnt 0" ::: "memory");

    // ---- O += P @ V ----
    bf16x16 ap  = afrag_bf16(pl, 32, lane, 0);
    bf16x16 bv0 = bfrag_bf16(Vt + 0 * 16 * 32, 32, lane, 0);
    bf16x16 bv1 = bfrag_bf16(Vt + 1 * 16 * 32, 32, lane, 0);
    bf16x16 bv2 = bfrag_bf16(Vt + 2 * 16 * 32, 32, lane, 0);
    bf16x16 bv3 = bfrag_bf16(Vt + 3 * 16 * 32, 32, lane, 0);
    o0 = WMMA_BF16(ap, bv0, o0);
    o1 = WMMA_BF16(ap, bv1, o1);
    o2 = WMMA_BF16(ap, bv2, o2);
    o3 = WMMA_BF16(ap, bv3, o3);
  }

  // ---- normalize + store attn out as [B, N, H*D] bf16 ----
#pragma unroll
  for (int r = 0; r < 8; ++r) {
    float inv = 1.f / lstate[r];
    int tok = q0 + r + 8 * g;
    __bf16* dst = ao + ((size_t)b * NN + tok) * CC + h * DD;
    dst[0 * 16 + nl] = (__bf16)(o0[r] * inv);
    dst[1 * 16 + nl] = (__bf16)(o1[r] * inv);
    dst[2 * 16 + nl] = (__bf16)(o2[r] * inv);
    dst[3 * 16 + nl] = (__bf16)(o3[r] * inv);
  }
}

// =====================================================================
// Kernel 3: output projection. ao[4096,1024]bf16 @ Wo[1024,1024]fp32 + bo
// -> fp32 d_out.
// =====================================================================
__global__ __launch_bounds__(256)
void k_oproj(const __bf16* __restrict__ ao, const float* __restrict__ Wo,
             const float* __restrict__ bo, float* __restrict__ out)
{
  __shared__ __bf16 ldsB[64 * 32];
  const int tid  = threadIdx.x;
  const int lane = tid & 31, wave = tid >> 5;
  const int n0 = blockIdx.x * 64;
  const int m0 = blockIdx.y * 128 + wave * 16;

  f32x8 acc0 = zero8(), acc1 = zero8(), acc2 = zero8(), acc3 = zero8();
  const int nc = tid & 63;
  const int kr = tid >> 6;

  for (int kk = 0; kk < CC; kk += 32) {
    __syncthreads();
#pragma unroll
    for (int i = 0; i < 8; ++i) {
      int krow = kr + i * 4;
      ldsB[nc * 32 + krow] = (__bf16)Wo[(size_t)(kk + krow) * CC + n0 + nc];
    }
    __syncthreads();
    bf16x16 a  = afrag_bf16(ao + (size_t)m0 * CC, CC, lane, kk);
    bf16x16 b0 = bfrag_bf16(ldsB + 0 * 16 * 32, 32, lane, 0);
    bf16x16 b1 = bfrag_bf16(ldsB + 1 * 16 * 32, 32, lane, 0);
    bf16x16 b2 = bfrag_bf16(ldsB + 2 * 16 * 32, 32, lane, 0);
    bf16x16 b3 = bfrag_bf16(ldsB + 3 * 16 * 32, 32, lane, 0);
    acc0 = WMMA_BF16(a, b0, acc0);
    acc1 = WMMA_BF16(a, b1, acc1);
    acc2 = WMMA_BF16(a, b2, acc2);
    acc3 = WMMA_BF16(a, b3, acc3);
  }

  const int g = lane >> 4, nl = lane & 15;
#pragma unroll
  for (int r = 0; r < 8; ++r) {
    int mrow = m0 + r + 8 * g;
#pragma unroll
    for (int sub = 0; sub < 4; ++sub) {
      int c = n0 + sub * 16 + nl;
      float val = (sub == 0) ? acc0[r] : (sub == 1) ? acc1[r]
                : (sub == 2) ? acc2[r] : acc3[r];
      out[(size_t)mrow * CC + c] = val + bo[c];
    }
  }
}

// =====================================================================
// Launch: qkv-proj -> attention -> out-proj, all on `stream`.
// Workspace layout (bf16): qh | kh | vt | ao, 4 x 4,194,304 elems = 32 MB.
// =====================================================================
extern "C" void kernel_launch(void* const* d_in, const int* in_sizes, int n_in,
                              void* d_out, int out_size, void* d_ws, size_t ws_size,
                              hipStream_t stream)
{
  (void)in_sizes; (void)n_in; (void)out_size; (void)ws_size;
  const float* qi = (const float*)d_in[0];
  const float* ki = (const float*)d_in[1];
  const float* vi = (const float*)d_in[2];
  const float* Wq = (const float*)d_in[3];
  const float* Wk = (const float*)d_in[4];
  const float* Wv = (const float*)d_in[5];
  const float* Wo = (const float*)d_in[6];
  const float* bo = (const float*)d_in[7];

  const size_t per = (size_t)BB * HH * NN * DD;   // 4,194,304 bf16 elems
  __bf16* qh = (__bf16*)d_ws;
  __bf16* kh = qh + per;
  __bf16* vt = kh + per;
  __bf16* ao = vt + per;

  dim3 g1(CC / 64, MTOT / 128, 3);
  k_qkv_proj<<<g1, 256, 0, stream>>>(qi, ki, vi, Wq, Wk, Wv, qh, kh, vt);

  dim3 g2(NN / 64, BB * HH);
  k_attn<<<g2, 128, 0, stream>>>(qh, kh, vt, ao);

  dim3 g3(CC / 64, MTOT / 128);
  k_oproj<<<g3, 256, 0, stream>>>(ao, Wo, bo, (float*)d_out);
}
